// GNN_87960930222107
// MI455X (gfx1250) — matrix operand order repaired and checked by reading, not compiled, not run on hardware
//
#include <hip/hip_runtime.h>
#include <hip/hip_bf16.h>

typedef __attribute__((ext_vector_type(16))) _Float16 v16h;
typedef __attribute__((ext_vector_type(8)))  float    v8f;

#define HCH   64
#define DA    300
#define DAP   320     // DA padded to multiple of 32 for mask-free WMMA fragments
#define DU    64

// ---------------------------------------------------------------------------
// WMMA helper (gfx1250, wave32): D = A(16x32 f16) x B(32x16 f16) + C(16x16 f32)
// ---------------------------------------------------------------------------
static __device__ __forceinline__ v8f wmma_f16(v16h a, v16h b, v8f c) {
  return __builtin_amdgcn_wmma_f32_16x16x32_f16(false, a, false, b, (short)0, c,
                                                false, false);
}

// Build a 16-half fragment from two 8-half contiguous runs (16B loads each).
// Works for global (global_load_b128) and LDS (ds_load_b128) pointers.
static __device__ __forceinline__ v16h frag_2x8(const _Float16* p0,
                                                const _Float16* p1) {
  union { v16h v; int4 q[2]; } u;
  u.q[0] = *reinterpret_cast<const int4*>(p0);
  u.q[1] = *reinterpret_cast<const int4*>(p1);
  return u.v;
}

// A fragment (16x32 f16) from row-major f16 [M x lda] (ISA 7.12.2 layout):
// lane<16: K = ks+0..7 and ks+16..23 ; lane>=16: K = ks+8..15 and ks+24..31.
static __device__ __forceinline__ v16h load_a_h(const _Float16* __restrict__ A,
                                                int row, int lda, int ks, int lane) {
  const int base = (lane < 16) ? 0 : 8;
  const _Float16* p = A + (size_t)row * lda + ks + base;
  return frag_2x8(p, p + 16);
}

// B fragment (32x16 f16) = W^T tile; W is [N x ldw] row-major f16.
// lane holds col n = n0+(lane&15), K = ks+(lane<16?0:8)+0..7 and ks+16+(lane<16?0:8)+0..7.
static __device__ __forceinline__ v16h load_b_h(const _Float16* __restrict__ W,
                                                int n0, int ldw, int ks, int lane) {
  const int n  = n0 + (lane & 15);
  const int kb = ks + ((lane < 16) ? 0 : 8);
  const _Float16* p = W + (size_t)n * ldw + kb;
  return frag_2x8(p, p + 16);
}

// A fragment from f32 source with compile-time-foldable K mask (K=300 tail).
static __device__ __forceinline__ v16h load_a_f(const float* __restrict__ A,
                                                int row, int lda, int ks, int lane, int K) {
  const int base = (lane < 16) ? 0 : 8;
  const float* p = A + (size_t)row * lda;
  v16h f;
#pragma unroll
  for (int j = 0; j < 8; ++j) {
    int k = ks + base + j;
    f[j] = (_Float16)((k < K) ? p[k] : 0.0f);
  }
#pragma unroll
  for (int j = 0; j < 8; ++j) {
    int k = ks + 16 + base + j;
    f[8 + j] = (_Float16)((k < K) ? p[k] : 0.0f);
  }
  return f;
}

// ---------------------------------------------------------------------------
// Weight pre-pack: f32 [N x K] -> f16 [N x Kpad], zero-padded K..Kpad.
// ---------------------------------------------------------------------------
__global__ void pack_w(const float* __restrict__ src, _Float16* __restrict__ dst,
                       int N, int K, int Kpad) {
  const int t = blockIdx.x * blockDim.x + threadIdx.x;
  if (t >= N * Kpad) return;
  const int n = t / Kpad, k = t - n * Kpad;
  dst[t] = (_Float16)((k < K) ? src[(size_t)n * K + k] : 0.0f);
}

// ---------------------------------------------------------------------------
// Y = relu(X[MxK] @ W16[64xKPAD]^T + b) -> f16 [M x 64]. One wave per 16x16 tile.
// ---------------------------------------------------------------------------
template <int K, int KPAD>
__global__ void proj_relu_gemm(const float* __restrict__ X,
                               const _Float16* __restrict__ W16,
                               const float* __restrict__ B,
                               _Float16* __restrict__ Y, int M) {
  const int lane = threadIdx.x & 31;
  const int gw   = blockIdx.x * (blockDim.x >> 5) + (threadIdx.x >> 5);
  const int tm = gw >> 2, tn = gw & 3;            // 64 cols = 4 N-tiles
  if (tm * 16 >= M) return;                       // wave-uniform guard
  const int m0 = tm * 16, n0 = tn * 16;
  v8f c = {};
  constexpr int NK = KPAD / 32;
#pragma unroll
  for (int kt = 0; kt < NK; ++kt) {
    v16h a = load_a_f(X, m0 + (lane & 15), K, kt * 32, lane, K);
    v16h b = load_b_h(W16, n0, KPAD, kt * 32, lane);
    c = wmma_f16(a, b, c);
  }
  const int n  = n0 + (lane & 15);
  const int r0 = m0 + ((lane < 16) ? 0 : 8);
  const float bias = B[n];
#pragma unroll
  for (int r = 0; r < 8; ++r) {
    float v = fmaxf(c[r] + bias, 0.0f);
    Y[(size_t)(r0 + r) * HCH + n] = (_Float16)v;
  }
}

// ---------------------------------------------------------------------------
// Fused SAGE update: D = opt_relu(Agg@WL^T + bl + Dst@WR^T). K = 64 (4 WMMAs).
// WL/WR (f16, 8KB each) are staged in LDS once per block; B frags via ds_load.
// ---------------------------------------------------------------------------
template <bool RELU, bool WRITE_H, bool WRITE_F>
__global__ void sage_gemm(const _Float16* __restrict__ Agg,
                          const _Float16* __restrict__ WL,
                          const float* __restrict__ BL,
                          const _Float16* __restrict__ Dst,
                          const _Float16* __restrict__ WR,
                          _Float16* __restrict__ outH,
                          float* __restrict__ outF, int M) {
  __shared__ _Float16 sWL[HCH * HCH];
  __shared__ _Float16 sWR[HCH * HCH];
  {
    const int4* gl = (const int4*)WL;
    const int4* gr = (const int4*)WR;
    int4* sl = (int4*)sWL;
    int4* sr = (int4*)sWR;
    for (int i = threadIdx.x; i < HCH * HCH / 8; i += blockDim.x) {
      sl[i] = gl[i];
      sr[i] = gr[i];
    }
  }
  __syncthreads();

  const int lane = threadIdx.x & 31;
  const int gw   = blockIdx.x * (blockDim.x >> 5) + (threadIdx.x >> 5);
  const int tm = gw >> 2, tn = gw & 3;
  if (tm * 16 >= M) return;
  const int m0 = tm * 16, n0 = tn * 16;
  const int rowA = m0 + (lane & 15);
  v8f c = {};
#pragma unroll
  for (int kt = 0; kt < 2; ++kt) {      // mean-aggregated neighbors @ WL^T
    v16h a = load_a_h(Agg, rowA, HCH, kt * 32, lane);
    v16h b = load_b_h(sWL, n0, HCH, kt * 32, lane);
    c = wmma_f16(a, b, c);
  }
#pragma unroll
  for (int kt = 0; kt < 2; ++kt) {      // dst features @ WR^T
    v16h a = load_a_h(Dst, rowA, HCH, kt * 32, lane);
    v16h b = load_b_h(sWR, n0, HCH, kt * 32, lane);
    c = wmma_f16(a, b, c);
  }
  const int n  = n0 + (lane & 15);
  const int r0 = m0 + ((lane < 16) ? 0 : 8);
  const float bias = BL[n];
#pragma unroll
  for (int r = 0; r < 8; ++r) {
    float v = c[r] + bias;
    if (RELU) v = fmaxf(v, 0.0f);
    const size_t idx = (size_t)(r0 + r) * HCH + n;
    if (WRITE_H) outH[idx] = (_Float16)v;
    if (WRITE_F) outF[idx] = v;
  }
}

// ---------------------------------------------------------------------------
// Edge scatter: agg[dst][c] += srcfeat[src][c]; deg[dst] += 1.  One thread per
// (edge, channel): 128B coalesced f16 gather per edge, f32 atomics out.
// ---------------------------------------------------------------------------
__global__ void scatter_edges(const _Float16* __restrict__ srcfeat,
                              const int* __restrict__ ei, int E,
                              float* __restrict__ agg, float* __restrict__ deg) {
  const long long t = (long long)blockIdx.x * blockDim.x + threadIdx.x;
  if (t >= (long long)E * HCH) return;
  const int e = (int)(t >> 6);
  const int c = (int)(t & 63);
  const int s = ei[e];          // row 0: src
  const int d = ei[E + e];      // row 1: dst
  atomicAdd(&agg[(size_t)d * HCH + c], (float)srcfeat[(size_t)s * HCH + c]);
  if (c == 0) atomicAdd(&deg[d], 1.0f);
}

// mean-normalize and convert to f16 (WMMA A-operand feed)
__global__ void mean_to_h(const float* __restrict__ agg,
                          const float* __restrict__ deg,
                          _Float16* __restrict__ out, int M) {
  const long long t = (long long)blockIdx.x * blockDim.x + threadIdx.x;
  if (t >= (long long)M * HCH) return;
  const int n = (int)(t >> 6);
  const float inv = 1.0f / fmaxf(deg[n], 1.0f);
  out[t] = (_Float16)(agg[t] * inv);
}

// head: out[m][o] = a2[m][:] . w_out[o][:] + b_out[o]   (OUT = 2)
__global__ void head_gemv(const float* __restrict__ a2,
                          const float* __restrict__ w_out,
                          const float* __restrict__ b_out,
                          float* __restrict__ out, int M) {
  const int t = blockIdx.x * blockDim.x + threadIdx.x;
  if (t >= M * 2) return;
  const int m = t >> 1, o = t & 1;
  const float* row = a2 + (size_t)m * HCH;
  const float* w   = w_out + o * HCH;
  float acc = b_out[o];
#pragma unroll
  for (int c = 0; c < HCH; ++c) acc += row[c] * w[c];
  out[t] = acc;
}

// ---------------------------------------------------------------------------
extern "C" void kernel_launch(void* const* d_in, const int* in_sizes, int n_in,
                              void* d_out, int out_size, void* d_ws, size_t ws_size,
                              hipStream_t stream) {
  const float* x_a    = (const float*)d_in[0];
  const float* x_u    = (const float*)d_in[1];
  const int*   ei_p   = (const int*)d_in[2];   // user -> article
  const int*   ei_b   = (const int*)d_in[3];   // article -> user
  const float* w_in_a = (const float*)d_in[4];
  const float* b_in_a = (const float*)d_in[5];
  const float* w_in_u = (const float*)d_in[6];
  const float* b_in_u = (const float*)d_in[7];
  const float* c1p_wl = (const float*)d_in[8];
  const float* c1p_bl = (const float*)d_in[9];
  const float* c1p_wr = (const float*)d_in[10];
  const float* c1b_wl = (const float*)d_in[11];
  const float* c1b_bl = (const float*)d_in[12];
  const float* c1b_wr = (const float*)d_in[13];
  const float* c2p_wl = (const float*)d_in[14];
  const float* c2p_bl = (const float*)d_in[15];
  const float* c2p_wr = (const float*)d_in[16];
  // d_in[17..19] = c2b_* : conv2 user branch is dropped by the reference head
  const float* w_out  = (const float*)d_in[20];
  const float* b_out  = (const float*)d_in[21];

  const int NA = in_sizes[0] / DA;
  const int NU = in_sizes[1] / DU;
  const int E  = in_sizes[2] / 2;

  // workspace carve-out (256B aligned)
  char* ws = (char*)d_ws;
  size_t off = 0;
  auto take = [&](size_t bytes) -> char* {
    char* p = ws + off;
    off = (off + bytes + 255) & ~(size_t)255;
    return p;
  };
  _Float16* aH    = (_Float16*)take((size_t)NA * HCH * 2);
  _Float16* uH    = (_Float16*)take((size_t)NU * HCH * 2);
  _Float16* a1H   = (_Float16*)take((size_t)NA * HCH * 2);
  _Float16* u1H   = (_Float16*)take((size_t)NU * HCH * 2);
  _Float16* aggAh = (_Float16*)take((size_t)NA * HCH * 2);
  _Float16* aggUh = (_Float16*)take((size_t)NU * HCH * 2);
  float*    aggA  = (float*)take((size_t)NA * HCH * 4);
  float*    aggU  = (float*)take((size_t)NU * HCH * 4);
  float*    degA  = (float*)take((size_t)NA * 4);
  float*    degU  = (float*)take((size_t)NU * 4);
  float*    a2    = (float*)take((size_t)NA * HCH * 4);
  // f16-packed weights
  _Float16* wA16   = (_Float16*)take((size_t)HCH * DAP * 2);   // zero-padded 300->320
  _Float16* wU16   = (_Float16*)take((size_t)HCH * HCH * 2);
  _Float16* w1pl16 = (_Float16*)take((size_t)HCH * HCH * 2);
  _Float16* w1pr16 = (_Float16*)take((size_t)HCH * HCH * 2);
  _Float16* w1bl16 = (_Float16*)take((size_t)HCH * HCH * 2);
  _Float16* w1br16 = (_Float16*)take((size_t)HCH * HCH * 2);
  _Float16* w2pl16 = (_Float16*)take((size_t)HCH * HCH * 2);
  _Float16* w2pr16 = (_Float16*)take((size_t)HCH * HCH * 2);

  const int TPB = 256;                          // 8 waves / block
  const unsigned gA  = (unsigned)(((NA / 16) * 4 + 7) / 8);   // GEMM tile waves
  const unsigned gU  = (unsigned)(((NU / 16) * 4 + 7) / 8);
  const long long tE = (long long)E * HCH;
  const unsigned gE  = (unsigned)((tE + TPB - 1) / TPB);
  const unsigned gNA = (unsigned)(((long long)NA * HCH + TPB - 1) / TPB);
  const unsigned gNU = (unsigned)(((long long)NU * HCH + TPB - 1) / TPB);
  const unsigned gW  = (unsigned)((HCH * DAP + TPB - 1) / TPB);
  const unsigned gW2 = (unsigned)((HCH * HCH + TPB - 1) / TPB);

  // 0) pre-pack weights to f16 (padded where needed)
  pack_w<<<gW,  TPB, 0, stream>>>(w_in_a, wA16,   HCH, DA,  DAP);
  pack_w<<<gW2, TPB, 0, stream>>>(w_in_u, wU16,   HCH, HCH, HCH);
  pack_w<<<gW2, TPB, 0, stream>>>(c1p_wl, w1pl16, HCH, HCH, HCH);
  pack_w<<<gW2, TPB, 0, stream>>>(c1p_wr, w1pr16, HCH, HCH, HCH);
  pack_w<<<gW2, TPB, 0, stream>>>(c1b_wl, w1bl16, HCH, HCH, HCH);
  pack_w<<<gW2, TPB, 0, stream>>>(c1b_wr, w1br16, HCH, HCH, HCH);
  pack_w<<<gW2, TPB, 0, stream>>>(c2p_wl, w2pl16, HCH, HCH, HCH);
  pack_w<<<gW2, TPB, 0, stream>>>(c2p_wr, w2pr16, HCH, HCH, HCH);

  // 1) per-type input projections (WMMA)
  proj_relu_gemm<DA, DAP><<<gA, TPB, 0, stream>>>(x_a, wA16, b_in_a, aH, NA);
  proj_relu_gemm<DU, DU ><<<gU, TPB, 0, stream>>>(x_u, wU16, b_in_u, uH, NU);

  // 2) conv1 aggregation (both edge types)
  hipMemsetAsync(aggA, 0, (size_t)NA * HCH * 4, stream);
  hipMemsetAsync(degA, 0, (size_t)NA * 4, stream);
  hipMemsetAsync(aggU, 0, (size_t)NU * HCH * 4, stream);
  hipMemsetAsync(degU, 0, (size_t)NU * 4, stream);
  scatter_edges<<<gE, TPB, 0, stream>>>(uH, ei_p, E, aggA, degA);
  scatter_edges<<<gE, TPB, 0, stream>>>(aH, ei_b, E, aggU, degU);
  mean_to_h<<<gNA, TPB, 0, stream>>>(aggA, degA, aggAh, NA);
  mean_to_h<<<gNU, TPB, 0, stream>>>(aggU, degU, aggUh, NU);

  // 3) conv1 fused SAGE GEMMs + ReLU (WMMA, LDS-staged weights)
  sage_gemm<true, true, false><<<gA, TPB, 0, stream>>>(aggAh, w1pl16, c1p_bl,
                                                       aH, w1pr16, a1H, nullptr, NA);
  sage_gemm<true, true, false><<<gU, TPB, 0, stream>>>(aggUh, w1bl16, c1b_bl,
                                                       uH, w1br16, u1H, nullptr, NU);

  // 4) conv2 (article branch only)
  hipMemsetAsync(aggA, 0, (size_t)NA * HCH * 4, stream);
  hipMemsetAsync(degA, 0, (size_t)NA * 4, stream);
  scatter_edges<<<gE, TPB, 0, stream>>>(u1H, ei_p, E, aggA, degA);
  mean_to_h<<<gNA, TPB, 0, stream>>>(aggA, degA, aggAh, NA);
  sage_gemm<false, false, true><<<gA, TPB, 0, stream>>>(aggAh, w2pl16, c2p_bl,
                                                        a1H, w2pr16, nullptr, a2, NA);

  // 5) output head
  head_gemv<<<(unsigned)((NA * 2 + TPB - 1) / TPB), TPB, 0, stream>>>(
      a2, w_out, b_out, (float*)d_out, NA);
}
